// Corr_33449205301665
// MI455X (gfx1250) — matrix operand order repaired and checked by reading, not compiled, not run on hardware
//
#include <hip/hip_runtime.h>
#include <hip/hip_bf16.h>

// ---------------------------------------------------------------------------
// Corr kernel for MI455X (gfx1250), wave32 + WMMA f16 16x16x32.
//
//   B=4, C=256, H=W=64, HW=4096, TOPK=3, alpha scalar input.
//   s[m,n] = alpha * (1 - gauss(m,n)) * <xn[:,m], xn[:,n]>     (symmetric)
//   xc[m,n] = exp(2 s[m,n]) / (R[m] * R[n]),  R[p] = sum_q exp(s[p,q])
//   out[b,k,n] = k-th largest over m of xc[m,n]
// ---------------------------------------------------------------------------

typedef __attribute__((ext_vector_type(16))) _Float16 v16h;
typedef __attribute__((ext_vector_type(8)))  _Float16 v8h;
typedef __attribute__((ext_vector_type(8)))  float    v8f;

#define HW   4096
#define CDIM 256
#define NB   4
// G = 1 / (2 * (64*0.05)^2) = 1/20.48 (same for rows/cols since H==W)
#define GINV   0.048828125f
// exp(-2*G) — constant ratio-of-ratios for the per-row gaussian recurrence
#define GQQ    0.90696066f

// ---------------------------------------------------------------------------
// Pass 0: channel L2-normalize, write transposed f16 panel Xt[b][pixel][c].
// One thread per pixel; channel reads are lane-coalesced (consecutive lanes ->
// consecutive pixels). x stays L2-hot (16MB) for the conversion re-read.
// ---------------------------------------------------------------------------
__global__ void corr_normalize(const float* __restrict__ x,
                               _Float16* __restrict__ Xt) {
  int gid = blockIdx.x * blockDim.x + threadIdx.x;   // 0..16383
  int b = gid >> 12;
  int p = gid & (HW - 1);
  const float* xp = x + (size_t)b * CDIM * HW + p;
  float ss = 0.0f;
#pragma unroll 8
  for (int c = 0; c < CDIM; ++c) {
    float v = xp[(size_t)c * HW];
    ss += v * v;
  }
  float scale = 1.0f / fmaxf(sqrtf(ss), 1e-12f);
  _Float16* o = Xt + ((size_t)b * HW + p) * CDIM;
#pragma unroll 4
  for (int i = 0; i < CDIM / 8; ++i) {
    v8h h;
#pragma unroll
    for (int j = 0; j < 8; ++j)
      h[j] = (_Float16)(xp[(size_t)(8 * i + j) * HW] * scale);
    *(v8h*)(o + 8 * i) = h;
  }
}

// branch-free insert of v into descending triple t[0]>=t[1]>=t[2]
__device__ __forceinline__ void top3_insert(float t[3], float v) {
  float m0 = fminf(t[0], v);
  t[0] = fmaxf(t[0], v);
  float m1 = fminf(t[1], m0);
  t[1] = fmaxf(t[1], m0);
  t[2] = fmaxf(t[2], m1);
}

// ---------------------------------------------------------------------------
// GEMM strip sweep. Block = 256 threads = 8 waves.
//   wave w: rowTile = w>>1 (4 x 16 rows), colHalf = w&1 (16 of 32 cols/step).
// The 4 row-tile waves re-read the same B tile -> absorbed by the WGP$,
// so no LDS staging needed; unique L2 traffic is ~512MB/pass vs 192MB L2
// holding all of Xt (8MB).
// PASS==1: accumulate R[row] = sum_col exp(s), write 1/R.
// PASS==2: rows are dest pixels n, cols are source pixels m (valid by
//          symmetry); track top-3 over m of exp(2s)*Rinv[m]*Rinv[n].
//
// WMMA 16-bit lane layouts (per CDNA5 ISA 7.12.2), served by contiguous
// 16B loads from Xt[pixel][c]:
//   A (16x32): lane l<16 row=l, elems 0..7 = K0..7,  8..15 = K16..23
//              lane l>=16 row=l-16, elems = K8..15 and K24..31
//   B (32x16): lane l<16 col=l, elems 0..7 = K0..7,  8..15 = K16..23
//              lane l>=16 col=l-16, elems = K8..15 and K24..31
// ---------------------------------------------------------------------------
template <int PASS>
__global__ void __launch_bounds__(256) corr_pass(
    const _Float16* __restrict__ Xt, const float* __restrict__ alphaPtr,
    float* __restrict__ recipSums, float* __restrict__ out) {
  __shared__ float sPart[8][16];      // PASS 1 cross-wave row sums
  __shared__ float sTop[4][16][3];    // PASS 2 cross-wave top3 merge

  const int b = blockIdx.y;
  const int strip = blockIdx.x;          // 64-row strip
  const int tid = threadIdx.x;
  const int lane = tid & 31;
  const int wave = tid >> 5;
  const int rowTile = wave >> 1;
  const int colHalf = wave & 1;
  const int li = lane & 15;
  const int hi = lane >> 4;              // K-half select
  // fold the factor 2 of pass2's exp(2s) into the alpha scale
  const float aScale = (PASS == 1 ? 1.0f : 2.0f) * alphaPtr[0];

  const size_t xtB = (size_t)b * HW * CDIM;
  const int rowBase = strip * 64 + rowTile * 16;
  const int rowLin = rowBase + hi * 8;   // first of this lane's 8 rows
  const int pmR = rowLin >> 6;           // image row: constant across r (no
  const int pmC = rowLin & 63;           // 64-boundary crossing since 8|rowLin)

  // ---- load A panel for this wave's 16 rows, all K (held in VGPRs) ----
  v16h A[8];
  {
    const _Float16* aRow = Xt + xtB + (size_t)(rowBase + li) * CDIM;
#pragma unroll
    for (int kt = 0; kt < 8; ++kt) {
      v8h lo = *(const v8h*)(aRow + kt * 32 + hi * 8);
      v8h hh = *(const v8h*)(aRow + kt * 32 + 16 + hi * 8);
#pragma unroll
      for (int i = 0; i < 8; ++i) { A[kt][i] = lo[i]; A[kt][8 + i] = hh[i]; }
    }
  }

  float rowAcc[8];
  float top3[8][3];
  float rowRecip[8];
#pragma unroll
  for (int r = 0; r < 8; ++r) {
    rowAcc[r] = 0.0f;
    top3[r][0] = top3[r][1] = top3[r][2] = 0.0f;
    rowRecip[r] = 0.0f;
  }
  if (PASS == 2) {
#pragma unroll
    for (int r = 0; r < 8; ++r)
      rowRecip[r] = recipSums[b * HW + rowLin + r];
  }

  // ---- sweep all 4096 columns, 32 per block step ----
  for (int cb = 0; cb < HW / 32; ++cb) {
    const int c0 = cb * 32 + colHalf * 16;
    const _Float16* bRow = Xt + xtB + (size_t)(c0 + li) * CDIM;

    v8f acc = {};
#pragma unroll
    for (int kt = 0; kt < 8; ++kt) {
      v8h blo = *(const v8h*)(bRow + kt * 32 + hi * 8);
      v8h bhh = *(const v8h*)(bRow + kt * 32 + 16 + hi * 8);
      v16h Bt;
#pragma unroll
      for (int i = 0; i < 8; ++i) { Bt[i] = blo[i]; Bt[8 + i] = bhh[i]; }
      acc = __builtin_amdgcn_wmma_f32_16x16x32_f16(
          false, A[kt], false, Bt, (short)0, acc, false, false);
    }
    if (cb + 1 < HW / 32)
      __builtin_prefetch(bRow + 32 * CDIM, 0, 1);   // next B tile (L2 resident)

    // ---- epilogue: D layout — VGPR r: row = r + hi*8, col = lane&15 ----
    const int pn = c0 + li;
    const int dr = pmR - (pn >> 6);     // constant across r
    const int d0 = pmC - (pn & 63);     // dc for r==0; dc_r = d0 + r
    float rm = 0.0f;
    if (PASS == 2) rm = recipSums[b * HW + pn];
    // gaussian by multiplicative recurrence: g_r = exp(-G*(dr^2+(d0+r)^2))
    //   g_{r+1} = g_r * q_r,  q_{r+1} = q_r * exp(-2G)
    // -> 2 exps + 2 fmuls/row instead of (imad+cvt+mul+exp)/row
    float g = __expf(-(float)(dr * dr + d0 * d0) * GINV);
    float q = __expf(-(float)(2 * d0 + 1) * GINV);
#pragma unroll
    for (int r = 0; r < 8; ++r) {
      const float t = aScale * acc[r];
      const float s = __builtin_fmaf(-t, g, t);     // t * (1 - g)
      if (PASS == 1) {
        rowAcc[r] += __expf(s);
      } else {
        top3_insert(top3[r], __expf(s) * rm * rowRecip[r]);
      }
      g *= q;
      q *= GQQ;
    }
  }

  if (PASS == 1) {
    // reduce over the 16 lanes holding each row, then across colHalf waves
#pragma unroll
    for (int r = 0; r < 8; ++r) {
      rowAcc[r] += __shfl_xor(rowAcc[r], 1, 32);
      rowAcc[r] += __shfl_xor(rowAcc[r], 2, 32);
      rowAcc[r] += __shfl_xor(rowAcc[r], 4, 32);
      rowAcc[r] += __shfl_xor(rowAcc[r], 8, 32);
    }
    if (li == 0) {
#pragma unroll
      for (int r = 0; r < 8; ++r) sPart[wave][hi * 8 + r] = rowAcc[r];
    }
    __syncthreads();
    if (tid < 64) {
      const int rt = tid >> 4, i = tid & 15;
      const float tot = sPart[2 * rt][i] + sPart[2 * rt + 1][i];
      recipSums[b * HW + strip * 64 + rt * 16 + i] = 1.0f / tot;
    }
  } else {
    // merge top3 across the 16 lanes of each half (butterfly)
#pragma unroll
    for (int m = 1; m <= 8; m <<= 1) {
#pragma unroll
      for (int r = 0; r < 8; ++r) {
        float o0 = __shfl_xor(top3[r][0], m, 32);
        float o1 = __shfl_xor(top3[r][1], m, 32);
        float o2 = __shfl_xor(top3[r][2], m, 32);
        top3_insert(top3[r], o0);
        top3_insert(top3[r], o1);
        top3_insert(top3[r], o2);
      }
    }
    if (colHalf == 0 && li == 0) {
#pragma unroll
      for (int r = 0; r < 8; ++r) {
        sTop[rowTile][hi * 8 + r][0] = top3[r][0];
        sTop[rowTile][hi * 8 + r][1] = top3[r][1];
        sTop[rowTile][hi * 8 + r][2] = top3[r][2];
      }
    }
    __syncthreads();
    if (colHalf == 1 && li == 0) {
#pragma unroll
      for (int r = 0; r < 8; ++r) {
        top3_insert(top3[r], sTop[rowTile][hi * 8 + r][0]);
        top3_insert(top3[r], sTop[rowTile][hi * 8 + r][1]);
        top3_insert(top3[r], sTop[rowTile][hi * 8 + r][2]);
        const int n = strip * 64 + rowTile * 16 + hi * 8 + r;
        out[((size_t)(b * 3 + 0) << 12) + n] = top3[r][0];
        out[((size_t)(b * 3 + 1) << 12) + n] = top3[r][1];
        out[((size_t)(b * 3 + 2) << 12) + n] = top3[r][2];
      }
    }
  }
}

extern "C" void kernel_launch(void* const* d_in, const int* in_sizes, int n_in,
                              void* d_out, int out_size, void* d_ws,
                              size_t ws_size, hipStream_t stream) {
  const float* x = (const float*)d_in[0];       // [4,256,64,64] f32
  const float* alpha = (const float*)d_in[1];   // scalar
  float* out = (float*)d_out;                   // [4,3,64,64] f32

  // workspace: Xt f16 [4][4096][256] (8 MB) | recipSums f32 [4][4096] (64 KB)
  _Float16* Xt = (_Float16*)d_ws;
  float* recipSums =
      (float*)((char*)d_ws + (size_t)NB * HW * CDIM * sizeof(_Float16));

  corr_normalize<<<NB * HW / 256, 256, 0, stream>>>(x, Xt);

  dim3 grid(HW / 64, NB);
  corr_pass<1><<<grid, 256, 0, stream>>>(Xt, alpha, recipSums, nullptr);
  corr_pass<2><<<grid, 256, 0, stream>>>(Xt, alpha, recipSums, out);
}
